// OLNRPN_71442486002219
// MI455X (gfx1250) — compile-verified, matrix operand order checked by reading, not run on hardware
//
#include <hip/hip_runtime.h>
#include <hip/hip_bf16.h>
#include <float.h>
#include <math.h>

typedef __attribute__((ext_vector_type(16))) _Float16 v16h;
typedef __attribute__((ext_vector_type(8)))  float    v8f;

#define NIMG   4
#define NLVL   5
#define MTOT   4768      // 1000+1000+1000+1000+768, == 298*16
#define WORDS  150       // 4768/32 = 149 used, +1 spare
#define TILES  298
#define POSTK  1000
#define NEGV   (-1.0e9f)
#define IOU_T  0.7f
#define SCLCLAMP 4.135166556742356f  /* log(1000/16) */
#define IMGSZ  1024.0f

__device__ __forceinline__ unsigned fkey(float x) {
  unsigned u = __float_as_uint(x);
  return u ^ ((u >> 31) ? 0xFFFFFFFFu : 0x80000000u);  // monotonic float->uint
}

// ---------------------------------------------------------------- zero mask
__global__ void zero_mask_kernel(unsigned* __restrict__ mask, long long nwords) {
  long long i = (long long)blockIdx.x * blockDim.x + threadIdx.x;
  long long stride = (long long)gridDim.x * blockDim.x;
  for (; i < nwords; i += stride) mask[i] = 0u;
}

// ------------------------------------------- Stage A: per (image,level) top-k
// radix-select threshold over logits, compact, bitonic-sort 1024 in LDS,
// decode boxes (apply_deltas + clip + min-size), write into concat arrays.
__global__ __launch_bounds__(1024) void stageA_topk_decode(
    const float* lg0, const float* lg1, const float* lg2, const float* lg3, const float* lg4,
    const float* dl0, const float* dl1, const float* dl2, const float* dl3, const float* dl4,
    const float* an0, const float* an1, const float* an2, const float* an3, const float* an4,
    float* __restrict__ cat_scores, float* __restrict__ cat_boxes) {
  const int hwa_t[5] = {196608, 49152, 12288, 3072, 768};
  const int k_t[5]   = {1000, 1000, 1000, 1000, 768};
  const int mb_t[5]  = {0, 1000, 2000, 3000, 4000};
  const float* lgs[5] = {lg0, lg1, lg2, lg3, lg4};
  const float* dls[5] = {dl0, dl1, dl2, dl3, dl4};
  const float* ans[5] = {an0, an1, an2, an3, an4};

  const int n = blockIdx.x / NLVL, lvl = blockIdx.x % NLVL;
  const int HWA = hwa_t[lvl], K = k_t[lvl], mbase = mb_t[lvl];
  const float* lg = lgs[lvl] + (size_t)n * HWA;
  const int tid = threadIdx.x;

  __shared__ unsigned hist[256];
  __shared__ unsigned s_pref;
  __shared__ int s_rem, cnt_acc, cnt_eq;
  __shared__ float s_sc[1024];
  __shared__ int   s_ix[1024];

  // 4-pass MSB->LSB radix select of the K-th largest key
  unsigned prefix = 0; int rem = K;
  for (int pass = 3; pass >= 0; --pass) {
    for (int b = tid; b < 256; b += 1024) hist[b] = 0u;
    __syncthreads();
    const unsigned sh = (unsigned)pass * 8u;
    const unsigned pmask = (pass == 3) ? 0u : (0xFFFFFFFFu << (sh + 8));
    for (int e = tid; e < HWA; e += 1024) {
      unsigned key = fkey(lg[e]);
      if ((key & pmask) == prefix) atomicAdd(&hist[(key >> sh) & 255u], 1u);
    }
    __syncthreads();
    if (tid == 0) {
      int acc = 0, b = 255;
      for (; b > 0; --b) { int c = (int)hist[b]; if (acc + c >= rem) break; acc += c; }
      s_pref = prefix | ((unsigned)b << sh);
      s_rem  = rem - acc;
    }
    __syncthreads();
    prefix = s_pref; rem = s_rem;
    __syncthreads();
  }
  const unsigned T = prefix;

  if (tid == 0) { cnt_acc = 0; cnt_eq = 0; }
  s_sc[tid] = -FLT_MAX; s_ix[tid] = 0x7FFFFFFF;
  __syncthreads();
  for (int e = tid; e < HWA; e += 1024) {
    float v = lg[e];
    unsigned key = fkey(v);
    int slot = -1;
    if (key > T) slot = atomicAdd(&cnt_acc, 1);
    else if (key == T) { int q = atomicAdd(&cnt_eq, 1); if (q < rem) slot = atomicAdd(&cnt_acc, 1); }
    if (slot >= 0) { s_sc[slot] = v; s_ix[slot] = e; }
  }
  __syncthreads();

  // bitonic sort 1024: score desc, tie -> lower index first (top_k stability)
  for (int k = 2; k <= 1024; k <<= 1)
    for (int j = k >> 1; j > 0; j >>= 1) {
      int ixj = tid ^ j;
      if (ixj > tid) {
        float sa = s_sc[tid], sb = s_sc[ixj];
        int   ia = s_ix[tid], ib = s_ix[ixj];
        bool up = ((tid & k) == 0);
        bool bf = (sa > sb) || (sa == sb && ia < ib);
        if (up != bf) { s_sc[tid] = sb; s_sc[ixj] = sa; s_ix[tid] = ib; s_ix[ixj] = ia; }
      }
      __syncthreads();
    }

  // decode selected boxes
  if (tid < K) {
    float sc = s_sc[tid];
    int idx = s_ix[tid];
    const float4 an = ((const float4*)ans[lvl])[idx];
    const float4 dl = ((const float4*)dls[lvl])[(size_t)n * HWA + idx];
    float wa = an.z - an.x, ha = an.w - an.y;
    float cxa = an.x + 0.5f * wa, cya = an.y + 0.5f * ha;
    float dw = fminf(dl.z, SCLCLAMP), dh = fminf(dl.w, SCLCLAMP);
    float cx = dl.x * wa + cxa, cy = dl.y * ha + cya;
    float w = wa * expf(dw), h = ha * expf(dh);
    float x0 = cx - 0.5f * w, y0 = cy - 0.5f * h, x1 = cx + 0.5f * w, y1 = cy + 0.5f * h;
    x0 = fminf(fmaxf(x0, 0.f), IMGSZ); y0 = fminf(fmaxf(y0, 0.f), IMGSZ);
    x1 = fminf(fmaxf(x1, 0.f), IMGSZ); y1 = fminf(fmaxf(y1, 0.f), IMGSZ);
    bool valid = ((x1 - x0) > 0.f) && ((y1 - y0) > 0.f);
    int m = mbase + tid;
    cat_scores[(size_t)n * MTOT + m] = valid ? sc : NEGV;
    float4 bx; bx.x = x0; bx.y = y0; bx.z = x1; bx.w = y1;
    ((float4*)cat_boxes)[(size_t)n * MTOT + m] = bx;
  }
}

// ------------- Stage B: per-image full sort of 4768 candidates (argsort desc)
// 8192-wide LDS bitonic sort (64KB LDS, fits the 320KB WGP LDS easily).
__global__ __launch_bounds__(1024) void stageB_sort(
    const float* __restrict__ cat_scores, const float* __restrict__ cat_boxes,
    float* __restrict__ sorted_scores, float* __restrict__ sorted_boxes,
    int* __restrict__ sorted_lvl) {
  const int n = blockIdx.x, tid = threadIdx.x;
  __shared__ float s_sc[8192];
  __shared__ int   s_ix[8192];
  for (int m = tid; m < 8192; m += 1024) {
    s_sc[m] = (m < MTOT) ? cat_scores[(size_t)n * MTOT + m] : -FLT_MAX;
    s_ix[m] = m;
  }
  __syncthreads();
  for (int k = 2; k <= 8192; k <<= 1)
    for (int j = k >> 1; j > 0; j >>= 1) {
      for (int t = tid; t < 4096; t += 1024) {
        int i = 2 * t - (t & (j - 1));
        int p = i + j;
        bool up = ((i & k) == 0);
        float sa = s_sc[i], sb = s_sc[p];
        int   ia = s_ix[i], ib = s_ix[p];
        bool bf = (sa > sb) || (sa == sb && ia < ib);
        if (up != bf) { s_sc[i] = sb; s_sc[p] = sa; s_ix[i] = ib; s_ix[p] = ia; }
      }
      __syncthreads();
    }
  for (int i = tid; i < MTOT; i += 1024) {
    int p = s_ix[i];
    sorted_scores[(size_t)n * MTOT + i] = s_sc[i];
    ((float4*)sorted_boxes)[(size_t)n * MTOT + i] =
        ((const float4*)cat_boxes)[(size_t)n * MTOT + p];
    int lvl = (p < 1000) ? 0 : (p < 2000) ? 1 : (p < 3000) ? 2 : (p < 4000) ? 3 : 4;
    sorted_lvl[(size_t)n * MTOT + i] = lvl;
  }
}

// ---- Stage C: 16x16 IoU tiles -> suppression bitmask.
// Same-level indicator via one-hot x one-hot V_WMMA_F32_16X16X32_F16 (exact 0/1
// arithmetic) replaces the reference's LVL_OFFSET coordinate trick.
// 1 wave per tile; wave32 ballot packs 2 mask rows (16 bits each) per reg.
__global__ __launch_bounds__(256) void stageC_mask(
    const float* __restrict__ sorted_boxes, const int* __restrict__ sorted_lvl,
    unsigned* __restrict__ mask) {
  const int wv = threadIdx.x >> 5, lane = threadIdx.x & 31;
  long long tile = (long long)blockIdx.x * 8 + wv;
  const long long total = (long long)NIMG * TILES * TILES;
  bool act = tile < total;
  long long t = act ? tile : 0;
  int n  = (int)(t / ((long long)TILES * TILES));
  int rc = (int)(t % ((long long)TILES * TILES));
  int rt = rc / TILES, ct = rc % TILES;
  act = act && (ct >= rt);   // strict upper triangle (diag tile handled per-elem)

  __shared__ float tA[8][16][4];
  __shared__ float tB[8][16][4];
  __shared__ int   lA[8][16], lB[8][16];
  const size_t base = (size_t)n * MTOT;
  if (lane < 16) {
    float4 b = ((const float4*)sorted_boxes)[base + rt * 16 + lane];
    tA[wv][lane][0] = b.x; tA[wv][lane][1] = b.y; tA[wv][lane][2] = b.z; tA[wv][lane][3] = b.w;
    lA[wv][lane] = sorted_lvl[base + rt * 16 + lane];
  } else {
    int c2 = lane - 16;
    float4 b = ((const float4*)sorted_boxes)[base + ct * 16 + c2];
    tB[wv][c2][0] = b.x; tB[wv][c2][1] = b.y; tB[wv][c2][2] = b.z; tB[wv][c2][3] = b.w;
    lB[wv][c2] = sorted_lvl[base + ct * 16 + c2];
  }
  __syncthreads();

  // Build one-hot A (16x32 f16, rows = levels of row-boxes) and B (32x16 f16,
  // cols = levels of col-boxes). Lanes>=16 carry K>=8 -> all zero.
  v16h A = {};
  v16h B = {};
  if (lane < 16) {
    int la = lA[wv][lane], lb = lB[wv][lane];
#pragma unroll
    for (int e = 0; e < 5; ++e) {
      A[e] = (la == e) ? (_Float16)1.0f : (_Float16)0.0f;  // A: elem e == K e (K<8)
      B[e] = (lb == e) ? (_Float16)1.0f : (_Float16)0.0f;  // B: elem e == K e (K<8)
    }
  }
  v8f c0 = {};
  // D[M][N] = 1.0 iff lvl(row M) == lvl(col N), else 0.0 (exact)
  v8f same = __builtin_amdgcn_wmma_f32_16x16x32_f16(
      false, A, false, B, (short)0, c0, false, false);

  const int half = lane >> 4, col = lane & 15;
  const float bx0 = tB[wv][col][0], by0 = tB[wv][col][1];
  const float bx1 = tB[wv][col][2], by1 = tB[wv][col][3];
  const float areaB = (bx1 - bx0) * (by1 - by0);
  const int jg = ct * 16 + col;
#pragma unroll
  for (int r = 0; r < 8; ++r) {
    const int mrow = r + 8 * half;               // C/D layout: M = r + 8*(lane/16)
    const float ax0 = tA[wv][mrow][0], ay0 = tA[wv][mrow][1];
    const float ax1 = tA[wv][mrow][2], ay1 = tA[wv][mrow][3];
    const float areaA = (ax1 - ax0) * (ay1 - ay0);
    float iw = fmaxf(fminf(ax1, bx1) - fmaxf(ax0, bx0), 0.f);
    float ih = fmaxf(fminf(ay1, by1) - fmaxf(ay0, by0), 0.f);
    float inter = iw * ih;
    float iou = inter / (areaA + areaB - inter + 1e-9f);
    const int ig = rt * 16 + mrow;
    bool pred = act && (same[r] > 0.5f) && (iou > IOU_T) && (jg > ig);
    unsigned bal = __builtin_amdgcn_ballot_w32(pred);
    if (lane == 0 && act) {
      const int wshift = (ct & 1) * 16;
      const int w = ct >> 1;
      unsigned lo = bal & 0xFFFFu, hi = bal >> 16;
      if (lo) atomicOr(&mask[((size_t)n * MTOT + rt * 16 + r)     * WORDS + w], lo << wshift);
      if (hi) atomicOr(&mask[((size_t)n * MTOT + rt * 16 + r + 8) * WORDS + w], hi << wshift);
    }
  }
}

// ------------- Stage D: sequential greedy NMS over the bitmask (per image).
// 150-word removed-bitmap in LDS; prefetch the next mask row to hide latency.
__global__ __launch_bounds__(160) void stageD_scan(
    const float* __restrict__ sorted_scores, const unsigned* __restrict__ mask,
    unsigned* __restrict__ keepArr) {
  const int n = blockIdx.x, tid = threadIdx.x;
  __shared__ unsigned removed[WORDS];
  __shared__ int s_keep;
  if (tid < WORDS) removed[tid] = 0u;
  __syncthreads();
  for (int i = 0; i < MTOT; ++i) {
    const unsigned* row = &mask[((size_t)n * MTOT + i) * WORDS];
    if (i + 1 < MTOT && tid < WORDS)
      __builtin_prefetch(&mask[((size_t)n * MTOT + i + 1) * WORDS + tid], 0, 1);
    if (tid == 0) {
      unsigned rm = removed[i >> 5];
      bool rb = (rm >> (i & 31)) & 1u;
      bool valid = sorted_scores[(size_t)n * MTOT + i] > -5.0e8f;  // > NEG/2
      int kp = (!rb && valid) ? 1 : 0;
      s_keep = kp;
      keepArr[(size_t)n * MTOT + i] = (unsigned)kp;
    }
    __syncthreads();
    if (s_keep && tid < WORDS) removed[tid] |= row[tid];
    __syncthreads();
  }
}

// ---- Stage E: replicate top_k(kept_scores, 1000): kept entries in score order,
// then suppressed/invalid entries (score==NEG ties) in index order.
__global__ __launch_bounds__(1024) void stageE_output(
    const float* __restrict__ sorted_scores, const float* __restrict__ sorted_boxes,
    const unsigned* __restrict__ keepArr, float* __restrict__ out) {
  const int n = blockIdx.x, tid = threadIdx.x;
  __shared__ int part[1024];
  const int CH = 5;  // 1024*5 >= MTOT
  int kf[CH];
  int s = 0;
#pragma unroll
  for (int q = 0; q < CH; ++q) {
    int m = tid * CH + q;
    kf[q] = (m < MTOT) ? (int)keepArr[(size_t)n * MTOT + m] : 0;
    s += kf[q];
  }
  part[tid] = s;
  __syncthreads();
  for (int off = 1; off < 1024; off <<= 1) {
    int add = (tid >= off) ? part[tid - off] : 0;
    __syncthreads();
    part[tid] += add;
    __syncthreads();
  }
  const int total = part[1023];
  int keptBefore = part[tid] - s;  // exclusive prefix of kept at chunk start
#pragma unroll
  for (int q = 0; q < CH; ++q) {
    int m = tid * CH + q;
    if (m < MTOT) {
      int pos = kf[q] ? keptBefore : total + (m - keptBefore);
      if (pos < POSTK) {
        float4 b = ((const float4*)sorted_boxes)[(size_t)n * MTOT + m];
        float sc = kf[q] ? sorted_scores[(size_t)n * MTOT + m] : NEGV;
        float* ob = &out[((size_t)n * POSTK + pos) * 4];
        ob[0] = b.x; ob[1] = b.y; ob[2] = b.z; ob[3] = b.w;
        out[(size_t)NIMG * POSTK * 4 + (size_t)n * POSTK + pos] = sc;
      }
      keptBefore += kf[q];
    }
  }
}

extern "C" void kernel_launch(void* const* d_in, const int* in_sizes, int n_in,
                              void* d_out, int out_size, void* d_ws, size_t ws_size,
                              hipStream_t stream) {
  (void)in_sizes; (void)n_in; (void)out_size; (void)ws_size;
  const float* lg[5]; const float* dl[5]; const float* an[5];
  for (int l = 0; l < 5; ++l) {
    lg[l] = (const float*)d_in[3 * l + 0];
    dl[l] = (const float*)d_in[3 * l + 1];
    an[l] = (const float*)d_in[3 * l + 2];
  }
  char* ws = (char*)d_ws;
  size_t off = 0;
  auto alloc = [&](size_t bytes) -> char* {
    char* p = ws + off;
    off = (off + bytes + 255) & ~(size_t)255;
    return p;
  };
  float*    cat_scores    = (float*)   alloc((size_t)NIMG * MTOT * sizeof(float));
  float*    cat_boxes     = (float*)   alloc((size_t)NIMG * MTOT * 4 * sizeof(float));
  float*    sorted_scores = (float*)   alloc((size_t)NIMG * MTOT * sizeof(float));
  float*    sorted_boxes  = (float*)   alloc((size_t)NIMG * MTOT * 4 * sizeof(float));
  int*      sorted_lvl    = (int*)     alloc((size_t)NIMG * MTOT * sizeof(int));
  unsigned* keepArr       = (unsigned*)alloc((size_t)NIMG * MTOT * sizeof(unsigned));
  unsigned* maskBuf       = (unsigned*)alloc((size_t)NIMG * MTOT * WORDS * sizeof(unsigned));

  const long long maskWords = (long long)NIMG * MTOT * WORDS;
  zero_mask_kernel<<<2048, 256, 0, stream>>>(maskBuf, maskWords);

  stageA_topk_decode<<<NIMG * NLVL, 1024, 0, stream>>>(
      lg[0], lg[1], lg[2], lg[3], lg[4],
      dl[0], dl[1], dl[2], dl[3], dl[4],
      an[0], an[1], an[2], an[3], an[4],
      cat_scores, cat_boxes);

  stageB_sort<<<NIMG, 1024, 0, stream>>>(cat_scores, cat_boxes,
                                         sorted_scores, sorted_boxes, sorted_lvl);

  const long long totalTiles = (long long)NIMG * TILES * TILES;
  const int blocksC = (int)((totalTiles + 7) / 8);
  stageC_mask<<<blocksC, 256, 0, stream>>>(sorted_boxes, sorted_lvl, maskBuf);

  stageD_scan<<<NIMG, 160, 0, stream>>>(sorted_scores, maskBuf, keepArr);

  stageE_output<<<NIMG, 1024, 0, stream>>>(sorted_scores, sorted_boxes, keepArr,
                                           (float*)d_out);
}
